// FraudGNN_15994458210355
// MI455X (gfx1250) — matrix operand order repaired and checked by reading, not compiled, not run on hardware
//
#include <hip/hip_runtime.h>
#include <hip/hip_bf16.h>

#define NNODES 100000
#define NEDGES 3200000
#define DF 128
#define KST 40   // padded LDS row stride (bf16 elems): 80B/row, 16B aligned, bank-conflict free

typedef __attribute__((ext_vector_type(16))) __bf16 v16bf;
typedef __attribute__((ext_vector_type(8)))  __bf16 v8bf;
typedef __attribute__((ext_vector_type(4)))  __bf16 v4bf;
typedef __attribute__((ext_vector_type(8)))  float  v8f;
typedef __attribute__((ext_vector_type(4)))  int    v4i;

#if defined(__HIP_DEVICE_COMPILE__) && __has_builtin(__builtin_amdgcn_global_load_async_to_lds_b128)
#define HAVE_ASYNC_LDS 1
#else
#define HAVE_ASYNC_LDS 0
#endif

// 16B global->LDS copy: async path (ASYNCcnt) on CDNA5, plain copy fallback
__device__ __forceinline__ void copy16(const __bf16* g, __bf16* l) {
#if HAVE_ASYNC_LDS
    typedef __attribute__((address_space(1))) v4i gv4i;
    typedef __attribute__((address_space(3))) v4i lv4i;
    __builtin_amdgcn_global_load_async_to_lds_b128(
        (gv4i*)const_cast<__bf16*>(g), (lv4i*)l, 0, 0);
#else
    *(uint4*)l = *(const uint4*)g;
#endif
}

__device__ __forceinline__ void async_wait() {
#if HAVE_ASYNC_LDS
#if __has_builtin(__builtin_amdgcn_s_wait_asynccnt)
    __builtin_amdgcn_s_wait_asynccnt(0);
#else
    asm volatile("s_wait_asynccnt 0" ::: "memory");
#endif
#endif
}

// ---------------------------------------------------------------- utilities

__global__ __launch_bounds__(256) void k_zero(float* __restrict__ p, int n) {
    int i = blockIdx.x * 256 + threadIdx.x;
    int stride = gridDim.x * 256;
    for (; i < n; i += stride) p[i] = 0.0f;
}

// f32 -> bf16 bulk convert (n4 = count/4)
__global__ __launch_bounds__(256) void k_cvt_bf16(const float* __restrict__ s,
                                                  __bf16* __restrict__ d, int n4) {
    int i = blockIdx.x * 256 + threadIdx.x;
    int stride = gridDim.x * 256;
    for (; i < n4; i += stride) {
        float4 v = *(const float4*)(s + (size_t)i * 4);
        v4bf b = { (__bf16)v.x, (__bf16)v.y, (__bf16)v.z, (__bf16)v.w };
        *(v4bf*)(d + (size_t)i * 4) = b;
    }
}

__global__ __launch_bounds__(256) void k_count(const long long* __restrict__ ei,
                                               float* __restrict__ cnt) {
    int e = blockIdx.x * 256 + threadIdx.x;
    if (e >= NEDGES) return;
    long long d = ei[(long long)NEDGES + e];
    atomicAdd(&cnt[d], 1.0f);
}

__global__ __launch_bounds__(256) void k_inv(float* __restrict__ cnt) {
    int i = blockIdx.x * 256 + threadIdx.x;
    if (i >= NNODES) return;
    float c = cnt[i];
    cnt[i] = 1.0f / fmaxf(c, 1.0f);
}

// one wave32 per edge: 128 bf16 feats = 4 per lane, HW float atomics into agg (L2-resident)
__global__ __launch_bounds__(256) void k_scatter_bf16(const __bf16* __restrict__ h,
                                                      const long long* __restrict__ ei,
                                                      float* __restrict__ agg) {
    int gid  = blockIdx.x * 256 + threadIdx.x;
    int e    = gid >> 5;
    int lane = gid & 31;
    if (e >= NEDGES) return;
    long long s = ei[e];
    long long d = ei[(long long)NEDGES + e];
    v4bf v = *(const v4bf*)(h + (size_t)s * DF + lane * 4);
    float* dst = agg + (size_t)d * DF + lane * 4;
    atomicAdd(dst + 0, (float)v[0]);
    atomicAdd(dst + 1, (float)v[1]);
    atomicAdd(dst + 2, (float)v[2]);
    atomicAdd(dst + 3, (float)v[3]);
}

// ---------------------------------------------------------------- WMMA GEMM

// A fragment (16x32 bf16, ISA layout): lane row = lane&15, K base = (lane>>4)*8;
// elements 0..7 at [kb, kb+8), 8..15 at [16+kb, 16+kb+8): two contiguous 16B chunks.
// B fragment (32x16 bf16): lane col = lane&15, K base = (lane>>4)*8, same two-chunk split as A.
__device__ __forceinline__ v16bf frag2(const __bf16* base, int o1, int o2) {
    v8bf lo = *(const v8bf*)(base + o1);
    v8bf hi = *(const v8bf*)(base + o2);
    v16bf r;
#pragma unroll
    for (int i = 0; i < 8; ++i) { r[i] = lo[i]; r[i + 8] = hi[i]; }
    return r;
}

__device__ __forceinline__ v8f wmma_bf16(v16bf a, v16bf b, v8f c) {
    return __builtin_amdgcn_wmma_f32_16x16x32_bf16(
        /*neg_a=*/false, a, /*neg_b=*/false, b,
        /*c_mod=*/(short)0, c, /*reuse_a=*/false, /*reuse_b=*/false);
}

// out = relu( (agg*inv) @ Wl^T + bl + x @ Wr^T );  optionally fused  logits = out @ Wc^T + bc
template <bool FUSE_CLS>
__global__ __launch_bounds__(256)
void sage_gemm(const __bf16* __restrict__ xb,     // [NNODES][128] bf16
               const float* __restrict__ agg,     // [NNODES][128] f32 (unscaled sums)
               const float* __restrict__ inv,     // [NNODES] 1/max(deg,1)
               const __bf16* __restrict__ Wlb,    // [128][128] bf16
               const float* __restrict__ bl,      // [128]
               const __bf16* __restrict__ Wrb,    // [128][128] bf16
               const float* __restrict__ Wc,      // [128]   (FUSE_CLS)
               const float* __restrict__ bc,      // [1]     (FUSE_CLS)
               __bf16* __restrict__ hout,         // [NNODES][128] (!FUSE_CLS)
               float* __restrict__ logits)        // [NNODES]      (FUSE_CLS)
{
    __shared__ __bf16 sX[DF * KST];
    __shared__ __bf16 sG[DF * KST];
    __shared__ __bf16 sWl[DF * KST];
    __shared__ __bf16 sWr[DF * KST];

    const int tid  = threadIdx.x;
    const int wave = tid >> 5;
    const int lane = tid & 31;
    const int m0   = blockIdx.x * DF;

    v8f acc[8];
#pragma unroll
    for (int nt = 0; nt < 8; ++nt)
#pragma unroll
        for (int i = 0; i < 8; ++i) acc[nt][i] = 0.0f;

    for (int kk = 0; kk < 4; ++kk) {           // K = 128 in 4 steps of 32
        // ---- stage slabs: x / Wl / Wr are pure 16B copies (async-to-LDS path) ----
        for (int t = tid; t < DF * 4; t += 256) {
            int row = t >> 2, c8 = t & 3;
            int rowg = m0 + row; if (rowg >= NNODES) rowg = NNODES - 1;
            copy16(xb + (size_t)rowg * DF + kk * 32 + c8 * 8, &sX[row * KST + c8 * 8]);
            copy16(Wlb + row * DF + kk * 32 + c8 * 8, &sWl[row * KST + c8 * 8]);
            copy16(Wrb + row * DF + kk * 32 + c8 * 8, &sWr[row * KST + c8 * 8]);
        }
        // ---- agg slab: per-row mean scale + f32->bf16 convert in registers ----
        for (int t = tid; t < DF * 8; t += 256) {
            int row = t >> 3, c4 = t & 7;
            int rowg = m0 + row; if (rowg >= NNODES) rowg = NNODES - 1;
            float s = inv[rowg];
            float4 v = *(const float4*)(agg + (size_t)rowg * DF + kk * 32 + c4 * 4);
            v4bf b = { (__bf16)(v.x * s), (__bf16)(v.y * s),
                       (__bf16)(v.z * s), (__bf16)(v.w * s) };
            *(v4bf*)(&sG[row * KST + c4 * 4]) = b;
        }
        async_wait();
        __syncthreads();

        // ---- compute: each wave owns 16 rows, all 128 output cols ----
        const int arow  = wave * 16 + (lane & 15);
        const int kb    = (lane >> 4) * 8;
        const int khalf = (lane >> 4) * 16;
        v16bf ax = frag2(&sX[arow * KST], kb, kb + 16);
        v16bf ag = frag2(&sG[arow * KST], kb, kb + 16);

        // software-pipelined B fragments: load nt+1 while nt's WMMAs execute
        v16bf bwl = frag2(&sWl[(lane & 15) * KST], kb, kb + 16);
        v16bf bwr = frag2(&sWr[(lane & 15) * KST], kb, kb + 16);
#pragma unroll
        for (int nt = 0; nt < 8; ++nt) {
            const int nxt = (nt < 7 ? nt + 1 : 7) * 16 + (lane & 15);
            v16bf nl = frag2(&sWl[nxt * KST], kb, kb + 16);
            v16bf nr = frag2(&sWr[nxt * KST], kb, kb + 16);
            acc[nt] = wmma_bf16(ag, bwl, acc[nt]);
            acc[nt] = wmma_bf16(ax, bwr, acc[nt]);
            bwl = nl; bwr = nr;
        }
        __syncthreads();
    }

    // ---- epilogue: bias + ReLU (+ fused classifier dot on layer 2) ----
    // C/D layout: acc[nt][r] is (row_local = (lane>>4)*8 + r, col = nt*16 + (lane&15))
    if (FUSE_CLS) {
        float pr[8];
#pragma unroll
        for (int r = 0; r < 8; ++r) pr[r] = 0.0f;
#pragma unroll
        for (int nt = 0; nt < 8; ++nt) {
            int col = nt * 16 + (lane & 15);
            float bias = bl[col];
            float wcv  = Wc[col];
#pragma unroll
            for (int r = 0; r < 8; ++r)
                pr[r] += fmaxf(acc[nt][r] + bias, 0.0f) * wcv;
        }
#pragma unroll
        for (int m = 1; m < 16; m <<= 1)
#pragma unroll
            for (int r = 0; r < 8; ++r) pr[r] += __shfl_xor(pr[r], m, 16);
        if ((lane & 15) == 0) {
            float bias = bc[0];
#pragma unroll
            for (int r = 0; r < 8; ++r) {
                int row = m0 + wave * 16 + (lane >> 4) * 8 + r;
                if (row < NNODES) logits[row] = pr[r] + bias;
            }
        }
    } else {
#pragma unroll
        for (int nt = 0; nt < 8; ++nt) {
            int col = nt * 16 + (lane & 15);
            float bias = bl[col];
#pragma unroll
            for (int r = 0; r < 8; ++r) {
                int row = m0 + wave * 16 + (lane >> 4) * 8 + r;
                if (row < NNODES)
                    hout[(size_t)row * DF + col] = (__bf16)fmaxf(acc[nt][r] + bias, 0.0f);
            }
        }
    }
}

// ---------------------------------------------------------------- launcher

extern "C" void kernel_launch(void* const* d_in, const int* in_sizes, int n_in,
                              void* d_out, int out_size, void* d_ws, size_t ws_size,
                              hipStream_t stream) {
    (void)in_sizes; (void)n_in; (void)out_size; (void)ws_size;
    const float*     x   = (const float*)d_in[0];
    const long long* ei  = (const long long*)d_in[1];
    const float*     Wl1 = (const float*)d_in[2];
    const float*     bl1 = (const float*)d_in[3];
    const float*     Wr1 = (const float*)d_in[4];
    const float*     Wl2 = (const float*)d_in[5];
    const float*     bl2 = (const float*)d_in[6];
    const float*     Wr2 = (const float*)d_in[7];
    const float*     Wc  = (const float*)d_in[8];
    const float*     bc  = (const float*)d_in[9];

    // workspace: agg f32 [N*128] | cnt/inv f32 [N] | h1 bf16 [N*128] | xb bf16 [N*128]
    //            | wb bf16 [4*128*128] (Wl1,Wr1,Wl2,Wr2)
    float*  agg = (float*)d_ws;
    float*  cnt = agg + (size_t)NNODES * DF;
    __bf16* h1  = (__bf16*)(cnt + NNODES);
    __bf16* xb  = h1 + (size_t)NNODES * DF;
    __bf16* wb  = xb + (size_t)NNODES * DF;
    __bf16* wl1b = wb;
    __bf16* wr1b = wb + 1 * DF * DF;
    __bf16* wl2b = wb + 2 * DF * DF;
    __bf16* wr2b = wb + 3 * DF * DF;

    const dim3 blk(256);
    const int zero1 = NNODES * DF + NNODES;           // agg + cnt (contiguous)
    const int zero2 = NNODES * DF;                    // agg only
    const int gZero1   = (zero1 + 255) / 256;
    const int gZero2   = (zero2 + 255) / 256;
    const int gCvtX    = (NNODES * DF / 4 + 255) / 256;
    const int gCvtW    = (DF * DF / 4 + 255) / 256;
    const int gCount   = (NEDGES + 255) / 256;
    const int gInv     = (NNODES + 255) / 256;
    const int gScatter = (NEDGES * 32 + 255) / 256;   // one wave per edge
    const int gGemm    = (NNODES + DF - 1) / DF;

    // ---- one-time bf16 casts (x + 4 weight matrices) ----
    k_cvt_bf16<<<gCvtX, blk, 0, stream>>>(x,   xb,   NNODES * DF / 4);
    k_cvt_bf16<<<gCvtW, blk, 0, stream>>>(Wl1, wl1b, DF * DF / 4);
    k_cvt_bf16<<<gCvtW, blk, 0, stream>>>(Wr1, wr1b, DF * DF / 4);
    k_cvt_bf16<<<gCvtW, blk, 0, stream>>>(Wl2, wl2b, DF * DF / 4);
    k_cvt_bf16<<<gCvtW, blk, 0, stream>>>(Wr2, wr2b, DF * DF / 4);

    // ---- layer 1 ----
    k_zero<<<gZero1, blk, 0, stream>>>(agg, zero1);
    k_count<<<gCount, blk, 0, stream>>>(ei, cnt);
    k_inv<<<gInv, blk, 0, stream>>>(cnt);             // cnt now holds 1/max(deg,1)
    k_scatter_bf16<<<gScatter, blk, 0, stream>>>(xb, ei, agg);
    sage_gemm<false><<<gGemm, blk, 0, stream>>>(
        xb, agg, cnt, wl1b, bl1, wr1b, nullptr, nullptr, h1, nullptr);

    // ---- layer 2 + fused classifier ----
    k_zero<<<gZero2, blk, 0, stream>>>(agg, zero2);
    k_scatter_bf16<<<gScatter, blk, 0, stream>>>(h1, ei, agg);
    sage_gemm<true><<<gGemm, blk, 0, stream>>>(
        h1, agg, cnt, wl2b, bl2, wr2b, Wc, bc, nullptr, (float*)d_out);
}
